// FTAttention_48000554500985
// MI455X (gfx1250) — compile-verified, matrix operand order checked by reading, not run on hardware
//
#include <hip/hip_runtime.h>
#include <math.h>

// FTAttention for MI455X (gfx1250, wave32, WMMA f32_16x16x32_f16).
// Pipeline (all on `stream`):
//   1. proj<FREQ>   : Q/K/V projections (f32 in -> f16 ws), fused bias,
//                     softmax scale folded into Q
//   2. attn_freq    : full softmax attention, L=128, hd=32 (async LDS staging)
//   3. outproj<FREQ>: O @ Wo^T + bo + v  -> d_out (fp32 "v1")
//   4. proj<TIME>   : projections over to_t(k), to_t(q), to_t(v1)
//   5. attn_time    : banded attention (window 10) -> 2 key tiles / q-tile
//   6. outproj<TIME>: O @ Wo^T + bo + v1 -> d_out (in-place add)
// Workspace: 4 regions x 32MB (Q,K,V,O staging in f16) = 128MB of d_ws.

#define NB 4
#define NC 64      // E (embed = channels)
#define NT 512
#define NF 128
#define NH 2
#define HD 32
#define WINL 10

typedef __attribute__((ext_vector_type(16))) _Float16 v16h;
typedef __attribute__((ext_vector_type(8)))  float    v8f;

#define WMMA_F16(a, b, c) \
  __builtin_amdgcn_wmma_f32_16x16x32_f16(false, (a), false, (b), (short)0, (c), false, false)

// ---- fragment gathers (wave32 layouts per CDNA5 ISA 7.12.2) ----------------
// A-frag / B-from-rows: element [i][k] = base[i*ld + k]
//   lane: i = lane&15 ; k pairs = {2u (u<4), 2u+8 (u>=4)} + 8*(lane>>4)
__device__ __forceinline__ v16h ld_frag(const _Float16* base, int ld) {
  const int lane = threadIdx.x & 31;
  const int i = lane & 15, hh = lane >> 4;
  v16h r;
#pragma unroll
  for (int u = 0; u < 8; ++u) {
    const int k0 = (u < 4 ? 2 * u : 2 * u + 8) + 8 * hh;
    r[2 * u]     = base[i * ld + k0];
    r[2 * u + 1] = base[i * ld + k0 + 1];
  }
  return r;
}
// Transposed source: element [i][k] = base[k*ld + i]  (B = V stored row=K)
__device__ __forceinline__ v16h ld_frag_t(const _Float16* base, int ld) {
  const int lane = threadIdx.x & 31;
  const int i = lane & 15, hh = lane >> 4;
  v16h r;
#pragma unroll
  for (int u = 0; u < 8; ++u) {
    const int k0 = (u < 4 ? 2 * u : 2 * u + 8) + 8 * hh;
    r[2 * u]     = base[k0 * ld + i];
    r[2 * u + 1] = base[(k0 + 1) * ld + i];
  }
  return r;
}

// ---- async global->LDS copy (gfx1250 ASYNCcnt path, bypasses VGPRs) --------
// GVS mode: mem = SADDR(64) + VADDR(32-bit byte offset); VDST = LDS byte addr.
__device__ __forceinline__ void async_ld_b64(unsigned lds_byte, unsigned gl_off,
                                             const void* sbase) {
  asm volatile("global_load_async_to_lds_b64 %0, %1, %2"
               :: "v"(lds_byte), "v"(gl_off),
                  "s"((unsigned long long)sbase)
               : "memory");
}
__device__ __forceinline__ void wait_async0() {
  asm volatile("s_wait_asynccnt 0" ::: "memory");
}
// AS(3) flat addresses carry the LDS byte offset in the low 32 bits.
__device__ __forceinline__ unsigned lds_off(const void* p) {
  return (unsigned)(unsigned long long)p;
}

// ---- 1/4: fused QKV projection --------------------------------------------
// MODE 0 (freq): rows = f (L=128), batch bt=b*NT+t.  MODE 1 (time): rows = t.
// NOTE reference swaps args: Q <- proj(k-input), K <- proj(q-input), V <- proj(v).
// Softmax scale (1/sqrt(hd)) is folded into the Q output here.
#define LDA 72   // padded LDS row stride (halves): 36 dwords -> conflict-free
template <int MODE>
__global__ void __launch_bounds__(128) proj_kernel(
    const float* __restrict__ xq, const float* __restrict__ xk,
    const float* __restrict__ xv, const float* __restrict__ Wi,
    const float* __restrict__ bi, _Float16* __restrict__ Qw,
    _Float16* __restrict__ Kw, _Float16* __restrict__ Vw) {
  __shared__ __align__(16) _Float16 Wl[192 * LDA];
  __shared__ float bl[192];
  __shared__ __align__(16) _Float16 At[3][16 * LDA];

  const int tid = threadIdx.x;
  const int blk = blockIdx.x;
  int b, t0, f0, bidx, L;
  if (MODE == 0) {                    // freq: 8 row-tiles of f per (b,t)
    const int bt = blk >> 3;
    b = bt / NT; t0 = bt % NT; f0 = (blk & 7) * 16;
    bidx = bt; L = NF;
  } else {                            // time: 32 row-tiles of t per (b,f)
    const int bf = blk >> 5;
    b = bf >> 7; f0 = bf & 127; t0 = (blk & 31) * 16;
    bidx = bf; L = NT;
  }

  for (int idx = tid; idx < 192 * 64; idx += 128)
    Wl[(idx >> 6) * LDA + (idx & 63)] = (_Float16)Wi[idx];
  for (int idx = tid; idx < 192; idx += 128) bl[idx] = bi[idx];

  const float* srcs[3] = {xq, xk, xv};
  const int r = tid & 15, e8 = tid >> 4;
#pragma unroll
  for (int p = 0; p < 3; ++p) {
    const float* s = srcs[p];
#pragma unroll
    for (int j = 0; j < 8; ++j) {
      const int e = e8 * 8 + j;
      size_t adr;
      if (MODE == 0) adr = ((size_t)(b * NC + e) * NT + t0) * NF + f0 + r;
      else           adr = ((size_t)(b * NC + e) * NT + t0 + r) * NF + f0;
      At[p][r * LDA + e] = (_Float16)s[adr];
    }
  }
  __syncthreads();

  const int w = tid >> 5, lane = tid & 31, n = lane & 15, hh = lane >> 4;
  for (int ct = w; ct < 12; ct += 4) {          // 12 col-tiles over 3E=192
    const int p = ct >> 2, o0 = ct * 16;
    v8f c = {};
#pragma unroll
    for (int kk = 0; kk < 2; ++kk) {
      v16h a  = ld_frag(&At[p][kk * 32], LDA);
      v16h bf = ld_frag(&Wl[o0 * LDA + kk * 32], LDA);
      c = WMMA_F16(a, bf, c);
    }
    _Float16* outp = (p == 0) ? Qw : (p == 1) ? Kw : Vw;
    const int oc = (o0 & 63) + n;               // col within this projection
    const int hidx = oc >> 5, d = oc & 31;
    const float bias = bl[o0 + n];
    const float scl = (p == 0) ? 0.17677669529663687f : 1.0f;  // 1/sqrt(32)
#pragma unroll
    for (int v = 0; v < 8; ++v) {
      const int l = (MODE == 0 ? f0 : t0) + v + 8 * hh;
      outp[(((size_t)bidx * NH + hidx) * L + l) * HD + d] =
          (_Float16)((c[v] + bias) * scl);
    }
  }
}

// ---- 2: frequency attention (L=128, hd=32, dense softmax) ------------------
#define LQK 36    // padded row stride for [*, 32] f16 tiles (72B, 8B-aligned)
#define LPF 136   // padded row stride for P [16,128]
__global__ void __launch_bounds__(256) attn_freq_kernel(
    const _Float16* __restrict__ Qw, const _Float16* __restrict__ Kw,
    const _Float16* __restrict__ Vw, _Float16* __restrict__ Ow) {
  __shared__ __align__(16) _Float16 Qs[NF * LQK], Ks[NF * LQK], Vs[NF * LQK];
  __shared__ __align__(16) _Float16 Pb[8][16 * LPF];

  const int bth = blockIdx.x;          // bt*NH + h
  const int tid = threadIdx.x;
  const _Float16* qg = Qw + (size_t)bth * NF * HD;
  const _Float16* kg = Kw + (size_t)bth * NF * HD;
  const _Float16* vg = Vw + (size_t)bth * NF * HD;

  // Async staging: 128 rows x 64B, as 8B chunks into padded rows.
  const unsigned qsB = lds_off(Qs), ksB = lds_off(Ks), vsB = lds_off(Vs);
  for (int c = tid; c < NF * 8; c += 256) {     // 8 chunks per 32-half row
    const int rr = c >> 3, cw = c & 7;
    const unsigned ldo = (unsigned)(rr * (LQK * 2) + cw * 8);
    const unsigned go  = (unsigned)(rr * 64 + cw * 8);
    async_ld_b64(qsB + ldo, go, qg);
    async_ld_b64(ksB + ldo, go, kg);
    async_ld_b64(vsB + ldo, go, vg);
  }
  wait_async0();
  __syncthreads();

  const int w = tid >> 5, lane = tid & 31, n = lane & 15, hh = lane >> 4;
  const v16h aQ = ld_frag(&Qs[16 * w * LQK], LQK);   // hd=32 -> single K-step
  v8f s[8];
#pragma unroll
  for (int jt = 0; jt < 8; ++jt) {
    v8f z = {};
    s[jt] = WMMA_F16(aQ, ld_frag(&Ks[16 * jt * LQK], LQK), z);
  }
  // softmax on C fragments: each lane-half holds rows v+8*hh, 1 col of 16
  _Float16* Pw = Pb[w];
#pragma unroll
  for (int v = 0; v < 8; ++v) {
    float mv = -1e30f;
#pragma unroll
    for (int jt = 0; jt < 8; ++jt) mv = fmaxf(mv, s[jt][v]);
    for (int m = 1; m < 16; m <<= 1) mv = fmaxf(mv, __shfl_xor(mv, m, 32));
    float sv = 0.f;
#pragma unroll
    for (int jt = 0; jt < 8; ++jt) { const float e = __expf(s[jt][v] - mv); s[jt][v] = e; sv += e; }
    for (int m = 1; m < 16; m <<= 1) sv += __shfl_xor(sv, m, 32);
    const float rs = 1.0f / sv;
#pragma unroll
    for (int jt = 0; jt < 8; ++jt)
      Pw[(v + 8 * hh) * LPF + jt * 16 + n] = (_Float16)(s[jt][v] * rs);
  }
  // O = P @ V  (same-wave LDS RAW: DS ops in-order per wave)
  const int bt = bth >> 1, h = bth & 1;
#pragma unroll
  for (int ct = 0; ct < 2; ++ct) {
    v8f o = {};
#pragma unroll
    for (int kk = 0; kk < 4; ++kk) {
      v16h a  = ld_frag(&Pw[kk * 32], LPF);
      v16h bv = ld_frag_t(&Vs[(kk * 32) * LQK + ct * 16], LQK);
      o = WMMA_F16(a, bv, o);
    }
#pragma unroll
    for (int v = 0; v < 8; ++v) {
      const int l = 16 * w + v + 8 * hh;
      Ow[((size_t)bt * NF + l) * NC + h * HD + ct * 16 + n] = (_Float16)o[v];
    }
  }
}

// ---- 5: time attention, banded (window WINL=10) ----------------------------
// Per 16-query tile, live keys span exactly local tiles [w, w+1] of the
// K/V window [q0-16, q0+128): 2 score WMMAs instead of 32.
#define LPT 40
__global__ void __launch_bounds__(256) attn_time_kernel(
    const _Float16* __restrict__ Qw, const _Float16* __restrict__ Kw,
    const _Float16* __restrict__ Vw, _Float16* __restrict__ Ow) {
  __shared__ __align__(16) _Float16 Qs[128 * LQK], Ks[144 * LQK], Vs[144 * LQK];
  __shared__ __align__(16) _Float16 Pb[8][16 * LPT];

  const int blk = blockIdx.x;                 // ((bf*NH + h)*4 + qb)
  const int qb = blk & 3, bfh = blk >> 2, h = bfh & 1, bf = bfh >> 1;
  const int q0 = qb * 128;
  const int tid = threadIdx.x;
  const _Float16* qg = Qw + (((size_t)bf * NH + h) * NT + q0) * HD;
  const _Float16* kg = Kw + ((size_t)bf * NH + h) * NT * HD;
  const _Float16* vg = Vw + ((size_t)bf * NH + h) * NT * HD;

  const unsigned qsB = lds_off(Qs), ksB = lds_off(Ks), vsB = lds_off(Vs);
  for (int c = tid; c < 128 * 8; c += 256) {
    const int rr = c >> 3, cw = c & 7;
    async_ld_b64(qsB + (unsigned)(rr * (LQK * 2) + cw * 8),
                 (unsigned)(rr * 64 + cw * 8), qg);
  }
  for (int c = tid; c < 144 * 8; c += 256) {  // K/V window [q0-16, q0+128)
    const int rr = c >> 3, cw = c & 7;
    const int j = q0 - 16 + rr;
    const unsigned ldo = (unsigned)(rr * (LQK * 2) + cw * 8);
    if (j < 0) {                              // zero-pad (fully masked rows)
      *(unsigned long long*)((char*)(void*)Ks + ldo) = 0ull;
      *(unsigned long long*)((char*)(void*)Vs + ldo) = 0ull;
    } else {
      const unsigned go = (unsigned)j * 64u + (unsigned)(cw * 8);
      async_ld_b64(ksB + ldo, go, kg);
      async_ld_b64(vsB + ldo, go, vg);
    }
  }
  wait_async0();
  __syncthreads();

  const int w = tid >> 5, lane = tid & 31, n = lane & 15, hh = lane >> 4;
  const int i0 = q0 + 16 * w;
  const v16h aQ = ld_frag(&Qs[16 * w * LQK], LQK);
  v8f s[2];
#pragma unroll
  for (int jj = 0; jj < 2; ++jj) {
    v8f z = {};
    s[jj] = WMMA_F16(aQ, ld_frag(&Ks[16 * (w + jj) * LQK], LQK), z);
  }
  const int jbase = q0 - 16 + 16 * w;         // global key of local col 0
  _Float16* Pw = Pb[w];
#pragma unroll
  for (int v = 0; v < 8; ++v) {
    const int i = i0 + v + 8 * hh;
    float mv = -1e30f;
#pragma unroll
    for (int jj = 0; jj < 2; ++jj) {
      const int j = jbase + jj * 16 + n;
      const bool ok = (j >= 0) && (j <= i) && (j > i - WINL);
      const float val = ok ? s[jj][v] : -1e30f;   // j==i always live
      s[jj][v] = val;
      mv = fmaxf(mv, val);
    }
    for (int m = 1; m < 16; m <<= 1) mv = fmaxf(mv, __shfl_xor(mv, m, 32));
    float sv = 0.f;
#pragma unroll
    for (int jj = 0; jj < 2; ++jj) { const float e = __expf(s[jj][v] - mv); s[jj][v] = e; sv += e; }
    for (int m = 1; m < 16; m <<= 1) sv += __shfl_xor(sv, m, 32);
    const float rs = 1.0f / sv;
#pragma unroll
    for (int jj = 0; jj < 2; ++jj)
      Pw[(v + 8 * hh) * LPT + jj * 16 + n] = (_Float16)(s[jj][v] * rs);
  }
#pragma unroll
  for (int ct = 0; ct < 2; ++ct) {
    v8f o = {};
    v16h a  = ld_frag(Pw, LPT);                               // K=32, one step
    v16h bv = ld_frag_t(&Vs[(16 * w) * LQK + ct * 16], LQK);  // local rows [16w,16w+32)
    o = WMMA_F16(a, bv, o);
#pragma unroll
    for (int v = 0; v < 8; ++v) {
      const int i = i0 + v + 8 * hh;
      Ow[((size_t)bf * NT + i) * NC + h * HD + ct * 16 + n] = (_Float16)o[v];
    }
  }
}

// ---- 3/6: output projection + bias + residual ------------------------------
template <int MODE>
__global__ void __launch_bounds__(128) outproj_kernel(
    const _Float16* __restrict__ Ow, const float* __restrict__ Wo,
    const float* __restrict__ bo, const float* resid, float* out) {
  __shared__ __align__(16) _Float16 Wl[64 * LDA];
  __shared__ float bl[64];
  __shared__ __align__(16) _Float16 At[16 * LDA];

  const int tid = threadIdx.x;
  for (int idx = tid; idx < 64 * 64; idx += 128)
    Wl[(idx >> 6) * LDA + (idx & 63)] = (_Float16)Wo[idx];
  if (tid < 64) bl[tid] = bo[tid];

  const int blk = blockIdx.x;
  const int L = (MODE == 0) ? NF : NT;
  const int tpb = L / 16;
  const int bidx = blk / tpb;                 // bt (freq) or bf (time)
  const int r0 = (blk % tpb) * 16;
  const _Float16* og = Ow + ((size_t)bidx * L + r0) * NC;
  for (int idx = tid; idx < 16 * 64; idx += 128)
    At[(idx >> 6) * LDA + (idx & 63)] = og[idx];

  const int w = tid >> 5, lane = tid & 31, n = lane & 15, hh = lane >> 4;
  const int n0 = 16 * w;
  const int cix = n0 + n;
  int b, t, f;
  if (MODE == 0) { b = bidx / NT; t = bidx % NT; f = r0; }
  else           { b = bidx >> 7; f = bidx & 127; t = r0; }
  // Warm L2 for the strided residual lines while the WMMAs run.
  {
    const int rr = 8 * hh;
    size_t pidx;
    if (MODE == 0) pidx = ((size_t)(b * NC + cix) * NT + t) * NF + f + rr;
    else           pidx = ((size_t)(b * NC + cix) * NT + t + rr) * NF + f;
    __builtin_prefetch(&resid[pidx], 0, 1);
  }
  __syncthreads();

  v8f c = {};
#pragma unroll
  for (int kk = 0; kk < 2; ++kk) {
    v16h a  = ld_frag(&At[kk * 32], LDA);
    v16h bw = ld_frag(&Wl[n0 * LDA + kk * 32], LDA);
    c = WMMA_F16(a, bw, c);
  }
  const float bias = bl[cix];
#pragma unroll
  for (int v = 0; v < 8; ++v) {
    const int rr = v + 8 * hh;
    size_t idx;
    if (MODE == 0) idx = ((size_t)(b * NC + cix) * NT + t) * NF + f + rr;
    else           idx = ((size_t)(b * NC + cix) * NT + t + rr) * NF + f;
    out[idx] = resid[idx] + c[v] + bias;
  }
}

// ---- host launcher ---------------------------------------------------------
extern "C" void kernel_launch(void* const* d_in, const int* in_sizes, int n_in,
                              void* d_out, int out_size, void* d_ws, size_t ws_size,
                              hipStream_t stream) {
  const float* k_in  = (const float*)d_in[0];
  const float* q_in  = (const float*)d_in[1];
  const float* v_in  = (const float*)d_in[2];
  const float* Wi_f  = (const float*)d_in[3];
  const float* bi_f  = (const float*)d_in[4];
  const float* Wo_f  = (const float*)d_in[5];
  const float* bo_f  = (const float*)d_in[6];
  const float* Wi_t  = (const float*)d_in[7];
  const float* bi_t  = (const float*)d_in[8];
  const float* Wo_t  = (const float*)d_in[9];
  const float* bo_t  = (const float*)d_in[10];
  float* out = (float*)d_out;

  // 4 x 32MB f16 staging regions (Q, K, V, O) reused by both stages.
  const size_t REG = (size_t)32 * 1024 * 1024;
  _Float16* Qp = (_Float16*)((char*)d_ws + 0 * REG);
  _Float16* Kp = (_Float16*)((char*)d_ws + 1 * REG);
  _Float16* Vp = (_Float16*)((char*)d_ws + 2 * REG);
  _Float16* Op = (_Float16*)((char*)d_ws + 3 * REG);
  (void)in_sizes; (void)n_in; (void)out_size; (void)ws_size;

  // -- frequency attention (query=k, key=q, value=v per reference) --
  proj_kernel<0><<<16384, 128, 0, stream>>>(k_in, q_in, v_in, Wi_f, bi_f, Qp, Kp, Vp);
  attn_freq_kernel<<<4096, 256, 0, stream>>>(Qp, Kp, Vp, Op);
  outproj_kernel<0><<<16384, 128, 0, stream>>>(Op, Wo_f, bo_f, v_in, out);  // out = v1

  // -- time attention (value = updated v1 = out), banded mask --
  proj_kernel<1><<<16384, 128, 0, stream>>>(k_in, q_in, out, Wi_t, bi_t, Qp, Kp, Vp);
  attn_time_kernel<<<4096, 256, 0, stream>>>(Qp, Kp, Vp, Op);
  outproj_kernel<1><<<16384, 128, 0, stream>>>(Op, Wo_t, bo_t, out, out);   // in-place add
}
